// GATBlock_29652454211795
// MI455X (gfx1250) — matrix operand order repaired and checked by reading, not compiled, not run on hardware
//
#include <hip/hip_runtime.h>

// ---------------------------------------------------------------------------
// GATv2 block for MI455X (gfx1250, wave32). GEMMs via v_wmma_f32_16x16x32_f16
// with LDS-staged B operands; graph phase via L2-resident gathers + f32 atomics.
// ---------------------------------------------------------------------------

typedef __attribute__((ext_vector_type(16))) _Float16 v16h;
typedef __attribute__((ext_vector_type(8)))  float    v8f;

#define LRELU_SLOPE 0.2f
#define LN_EPS      1e-5f

// Convert 8 f32 (two float4) into elements [B..B+7] of a v16h A-fragment.
template <int B>
__device__ __forceinline__ void cvt8(v16h& a, float4 p, float4 q) {
  a[B + 0] = (_Float16)p.x; a[B + 1] = (_Float16)p.y;
  a[B + 2] = (_Float16)p.z; a[B + 3] = (_Float16)p.w;
  a[B + 4] = (_Float16)q.x; a[B + 5] = (_Float16)q.y;
  a[B + 6] = (_Float16)q.z; a[B + 7] = (_Float16)q.w;
}

// Ordered-uint encoding for float atomic max.
__device__ __forceinline__ unsigned fkey(float f) {
  unsigned u = __float_as_uint(f);
  return u ^ ((unsigned)((int)u >> 31) | 0x80000000u);
}
__device__ __forceinline__ float funkey(unsigned k) {
  unsigned u = (k & 0x80000000u) ? (k ^ 0x80000000u) : ~k;
  return __uint_as_float(u);
}

// ---------------------------------------------------------------------------
// Pack a row-major f32 weight W[K x Ncols] into f16 WMMA-B fragments.
// Fragment layout (per 16x32-K x 16-N tile): 32 lanes x 16 halves, contiguous.
// Within a K-tile: K = (lane<16 ? 0 : 16) + e  (contiguous k order, same as the A loads), N = lane&15.
// Fragment index = nt*ktiles + kt.
// ---------------------------------------------------------------------------
__global__ __launch_bounds__(256) void k_pack_b(const float* __restrict__ W,
                                                _Float16* __restrict__ Wp,
                                                int K, int ncols) {
  int t = blockIdx.x * 256 + threadIdx.x;
  int ktiles = K >> 5;
  int total = ktiles * (ncols >> 4) * 512;
  if (t >= total) return;
  int e    = t & 15;
  int lane = (t >> 4) & 31;
  int frag = t >> 9;
  int nt = frag / ktiles;
  int kt = frag - nt * ktiles;
  int kk = kt * 32 + ((lane < 16) ? 0 : 16) + e;
  int cc = nt * 16 + (lane & 15);
  Wp[t] = (_Float16)W[kk * ncols + cc];
}

// ---------------------------------------------------------------------------
// Init: zero scatter target, denominators, max-keys, stat scalars.
// ---------------------------------------------------------------------------
__global__ __launch_bounds__(256) void k_init(float* __restrict__ gat,
                                              float* __restrict__ denom,
                                              unsigned* __restrict__ nmax,
                                              float* __restrict__ sums, int n) {
  int i = blockIdx.x * 256 + threadIdx.x;
  if (i < n * 128) gat[i] = 0.0f;
  if (i < n * 2) { denom[i] = 0.0f; nmax[i] = 0u; }
  if (i < 4) sums[i] = 0.0f;
}

// ---------------------------------------------------------------------------
// Fused GEMM: xl = x@W_l + b_l, xr = x@W_r + b_r.  x:[n,64], out:[npad,128].
// Block = 256 threads = 8 waves; each wave owns 16 rows, all 8 N-tiles.
// Packed B matrices staged into LDS once per block (32 KB of 320 KB/WGP).
// Output buffers are padded to a multiple of 128 rows -> unguarded stores.
// ---------------------------------------------------------------------------
__global__ __launch_bounds__(256) void k_gemm_xlxr(
    const float* __restrict__ x,
    const _Float16* __restrict__ Wlp, const _Float16* __restrict__ Wrp,
    const float* __restrict__ b_l, const float* __restrict__ b_r,
    float* __restrict__ xl, float* __restrict__ xr, int n) {
  __shared__ _Float16 sW[16384];          // [0..8191]=W_l frags, [8192..]=W_r frags
  {
    uint4* d = (uint4*)sW;
    const uint4* s0 = (const uint4*)Wlp;
    const uint4* s1 = (const uint4*)Wrp;
    for (int i = threadIdx.x; i < 1024; i += 256) {
      d[i] = s0[i];
      d[1024 + i] = s1[i];
    }
  }
  __syncthreads();
  const _Float16* sWl = sW;
  const _Float16* sWr = sW + 8192;

  int lane    = threadIdx.x & 31;
  int wave    = threadIdx.x >> 5;
  int rowbase = blockIdx.x * 128 + wave * 16;
  int m       = lane & 15;
  int kbase   = (lane < 16) ? 0 : 16;     // contiguous k order (matches packed B)
  int row     = rowbase + m;
  if (row >= n) row = n - 1;              // clamp loads to real input
  const float* xrow = x + row * 64;

  v16h a0, a1;
  {
    const float* p = xrow + kbase;        // K half 0: K in [0,32)
    cvt8<0>(a0, *(const float4*)p,        *(const float4*)(p + 4));
    cvt8<8>(a0, *(const float4*)(p + 8), *(const float4*)(p + 12));
  }
  {
    const float* p = xrow + 32 + kbase;   // K half 1: K in [32,64)
    cvt8<0>(a1, *(const float4*)p,        *(const float4*)(p + 4));
    cvt8<8>(a1, *(const float4*)(p + 8), *(const float4*)(p + 12));
  }

  int col    = lane & 15;
  int rstore = rowbase + ((lane >= 16) ? 8 : 0);   // f32 C layout

  #pragma unroll
  for (int nt = 0; nt < 8; ++nt) {
    v16h bl0 = *(const v16h*)(sWl + ((nt * 2 + 0) * 32 + lane) * 16);
    v16h bl1 = *(const v16h*)(sWl + ((nt * 2 + 1) * 32 + lane) * 16);
    v16h br0 = *(const v16h*)(sWr + ((nt * 2 + 0) * 32 + lane) * 16);
    v16h br1 = *(const v16h*)(sWr + ((nt * 2 + 1) * 32 + lane) * 16);
    v8f cl = {};
    v8f cr = {};
    cl = __builtin_amdgcn_wmma_f32_16x16x32_f16(false, a0, false, bl0, (short)0, cl, false, false);
    cl = __builtin_amdgcn_wmma_f32_16x16x32_f16(false, a1, false, bl1, (short)0, cl, false, false);
    cr = __builtin_amdgcn_wmma_f32_16x16x32_f16(false, a0, false, br0, (short)0, cr, false, false);
    cr = __builtin_amdgcn_wmma_f32_16x16x32_f16(false, a1, false, br1, (short)0, cr, false, false);
    float bbl = b_l[nt * 16 + col];
    float bbr = b_r[nt * 16 + col];
    #pragma unroll
    for (int r = 0; r < 8; ++r) {
      int rr = rstore + r;
      xl[rr * 128 + nt * 16 + col] = cl[r] + bbl;   // padded buffers: no guard
      xr[rr * 128 + nt * 16 + col] = cr[r] + bbr;
    }
  }
}

// ---------------------------------------------------------------------------
// Edge attention logits: leaky_relu(xl[src]+xr[dst]+ea@W_e) . att[h],
// plus ordered-uint atomic max per (dst, head). One thread per (edge, head).
// ---------------------------------------------------------------------------
__global__ __launch_bounds__(256) void k_logits(
    const float* __restrict__ xl, const float* __restrict__ xr,
    const long long* __restrict__ ei, const float* __restrict__ eattr,
    const float* __restrict__ W_e, const float* __restrict__ att,
    float* __restrict__ logits, unsigned* __restrict__ nmax, int E_, int n) {
  __shared__ float sWe[512];   // [4][128]
  __shared__ float sAtt[128];  // [2][64]
  for (int i = threadIdx.x; i < 512; i += 256) sWe[i] = W_e[i];
  if (threadIdx.x < 128) sAtt[threadIdx.x] = att[threadIdx.x];
  __syncthreads();

  int tt = blockIdx.x * 256 + threadIdx.x;
  if (tt >= (E_ + n) * 2) return;
  int e = tt >> 1, h = tt & 1;

  int src, dst;
  float a0, a1, a2, a3;
  if (e < E_) {
    src = (int)ei[e];
    dst = (int)ei[E_ + e];
    const float* ea = eattr + (size_t)e * 4;
    a0 = ea[0]; a1 = ea[1]; a2 = ea[2]; a3 = ea[3];
  } else {
    src = dst = e - E_;           // self loop, edge_attr = 0
    a0 = a1 = a2 = a3 = 0.0f;
  }

  const float* pl = xl + (size_t)src * 128 + h * 64;
  const float* pr = xr + (size_t)dst * 128 + h * 64;
  __builtin_prefetch(pr, 0, 1);   // global_prefetch into L2

  float acc = 0.0f;
  #pragma unroll
  for (int f = 0; f < 64; f += 4) {
    float4 vl = *(const float4*)(pl + f);
    float4 vr = *(const float4*)(pr + f);
    float lv[4] = {vl.x, vl.y, vl.z, vl.w};
    float rv[4] = {vr.x, vr.y, vr.z, vr.w};
    #pragma unroll
    for (int j = 0; j < 4; ++j) {
      int c = h * 64 + f + j;
      float s = lv[j] + rv[j] +
                a0 * sWe[c] + a1 * sWe[128 + c] + a2 * sWe[256 + c] + a3 * sWe[384 + c];
      s = (s > 0.0f) ? s : LRELU_SLOPE * s;
      acc += s * sAtt[c];
    }
  }
  logits[tt] = acc;
  atomicMax(&nmax[dst * 2 + h], fkey(acc));
}

// ---------------------------------------------------------------------------
// ex = exp(logit - max[dst]); denom[dst] += ex (segment softmax, pass 2).
// ---------------------------------------------------------------------------
__global__ __launch_bounds__(256) void k_expsum(
    const long long* __restrict__ ei, float* __restrict__ logits,
    const unsigned* __restrict__ nmax, float* __restrict__ denom, int E_, int n) {
  int tt = blockIdx.x * 256 + threadIdx.x;
  if (tt >= (E_ + n) * 2) return;
  int e = tt >> 1, h = tt & 1;
  int dst = (e < E_) ? (int)ei[E_ + e] : (e - E_);
  float m  = funkey(nmax[dst * 2 + h]);
  float ex = __expf(logits[tt] - m);
  logits[tt] = ex;
  atomicAdd(&denom[dst * 2 + h], ex);
}

// ---------------------------------------------------------------------------
// Scatter: gat[dst] += alpha * xl[src] (per head, float atomics).
// ---------------------------------------------------------------------------
__global__ __launch_bounds__(256) void k_scatter(
    const long long* __restrict__ ei, const float* __restrict__ logits,
    const float* __restrict__ denom, const float* __restrict__ xl,
    float* __restrict__ gat, int E_, int n) {
  int tt = blockIdx.x * 256 + threadIdx.x;
  if (tt >= (E_ + n) * 2) return;
  int e = tt >> 1, h = tt & 1;
  int src, dst;
  if (e < E_) { src = (int)ei[e]; dst = (int)ei[E_ + e]; }
  else        { src = dst = e - E_; }
  float alpha = logits[tt] / denom[dst * 2 + h];
  const float* pl = xl + (size_t)src * 128 + h * 64;
  float* po = gat + (size_t)dst * 128 + h * 64;
  #pragma unroll
  for (int f = 0; f < 64; f += 4) {
    float4 v = *(const float4*)(pl + f);
    atomicAdd(po + f + 0, v.x * alpha);
    atomicAdd(po + f + 1, v.y * alpha);
    atomicAdd(po + f + 2, v.z * alpha);
    atomicAdd(po + f + 3, v.w * alpha);
  }
}

// ---------------------------------------------------------------------------
// gat += bias (in place) and accumulate global sum / sumsq -> sums[0..1].
// ---------------------------------------------------------------------------
__global__ __launch_bounds__(256) void k_biasstats(
    float* __restrict__ gat, const float* __restrict__ bias, int total,
    float* __restrict__ sums) {
  float s = 0.0f, s2 = 0.0f;
  for (int i = blockIdx.x * 256 + threadIdx.x; i < total; i += gridDim.x * 256) {
    float v = gat[i] + bias[i & 127];
    gat[i] = v;
    s += v; s2 += v * v;
  }
  __shared__ float sh[256], sh2[256];
  sh[threadIdx.x] = s; sh2[threadIdx.x] = s2;
  __syncthreads();
  for (int o = 128; o > 0; o >>= 1) {
    if (threadIdx.x < (unsigned)o) {
      sh[threadIdx.x]  += sh[threadIdx.x + o];
      sh2[threadIdx.x] += sh2[threadIdx.x + o];
    }
    __syncthreads();
  }
  if (threadIdx.x == 0) { atomicAdd(&sums[0], sh[0]); atomicAdd(&sums[1], sh2[0]); }
}

// Plain global sum / sumsq -> out2[0..1].
__global__ __launch_bounds__(256) void k_stats(
    const float* __restrict__ p, int total, float* __restrict__ out2) {
  float s = 0.0f, s2 = 0.0f;
  for (int i = blockIdx.x * 256 + threadIdx.x; i < total; i += gridDim.x * 256) {
    float v = p[i];
    s += v; s2 += v * v;
  }
  __shared__ float sh[256], sh2[256];
  sh[threadIdx.x] = s; sh2[threadIdx.x] = s2;
  __syncthreads();
  for (int o = 128; o > 0; o >>= 1) {
    if (threadIdx.x < (unsigned)o) {
      sh[threadIdx.x]  += sh[threadIdx.x + o];
      sh2[threadIdx.x] += sh2[threadIdx.x + o];
    }
    __syncthreads();
  }
  if (threadIdx.x == 0) { atomicAdd(&out2[0], sh[0]); atomicAdd(&out2[1], sh2[0]); }
}

// ---------------------------------------------------------------------------
// Graph layernorm 1 + relu -> hbuf  (stats in sums[0..1], feature = i&127).
// ---------------------------------------------------------------------------
__global__ __launch_bounds__(256) void k_ln1relu(
    const float* __restrict__ gat, const float* __restrict__ w,
    const float* __restrict__ b, const float* __restrict__ sums,
    float* __restrict__ hbuf, int total, int totalpad) {
  int i = blockIdx.x * 256 + threadIdx.x;
  if (i >= totalpad) return;
  float cnt  = (float)total;
  float mean = sums[0] / cnt;
  float var  = sums[1] / cnt - mean * mean;
  float inv  = 1.0f / (sqrtf(fmaxf(var, 0.0f)) + LN_EPS);
  int c = i & 127;
  float g = (i < total) ? gat[i] : 0.0f;   // zero-fill pad rows for the GEMM
  float v = (g - mean) * inv * w[c] + b[c];
  hbuf[i] = (v > 0.0f) ? v : 0.0f;
}

// ---------------------------------------------------------------------------
// GEMM: y = hbuf @ W_lin + b_lin.  hbuf:[npad,128], y:[npad,64]. K=128.
// Packed B staged to LDS (16 KB); unguarded padded stores.
// ---------------------------------------------------------------------------
__global__ __launch_bounds__(256) void k_gemm_lin(
    const float* __restrict__ hbuf, const _Float16* __restrict__ Wp,
    const float* __restrict__ blin, float* __restrict__ y, int npad) {
  __shared__ _Float16 sW[8192];
  {
    uint4* d = (uint4*)sW;
    const uint4* s0 = (const uint4*)Wp;
    for (int i = threadIdx.x; i < 1024; i += 256) d[i] = s0[i];
  }
  __syncthreads();

  int lane    = threadIdx.x & 31;
  int wave    = threadIdx.x >> 5;
  int rowbase = blockIdx.x * 128 + wave * 16;
  int m       = lane & 15;
  int kbase   = (lane < 16) ? 0 : 16;
  int row     = rowbase + m;              // hbuf is padded: safe
  const float* hr = hbuf + row * 128;

  v16h a[4];
  #pragma unroll
  for (int kt = 0; kt < 4; ++kt) {
    const float* p = hr + kt * 32 + kbase;
    cvt8<0>(a[kt], *(const float4*)p,        *(const float4*)(p + 4));
    cvt8<8>(a[kt], *(const float4*)(p + 8), *(const float4*)(p + 12));
  }

  int col    = lane & 15;
  int rstore = rowbase + ((lane >= 16) ? 8 : 0);

  #pragma unroll
  for (int nt = 0; nt < 4; ++nt) {
    v8f c = {};
    #pragma unroll
    for (int kt = 0; kt < 4; ++kt) {
      v16h bfrag = *(const v16h*)(sW + ((nt * 4 + kt) * 32 + lane) * 16);
      c = __builtin_amdgcn_wmma_f32_16x16x32_f16(false, a[kt], false, bfrag, (short)0, c, false, false);
    }
    float bb = blin[nt * 16 + col];
    #pragma unroll
    for (int r = 0; r < 8; ++r) {
      int rr = rstore + r;
      y[rr * 64 + nt * 16 + col] = c[r] + bb;   // padded: no guard
    }
  }
  (void)npad;
}

// ---------------------------------------------------------------------------
// Graph layernorm 2 + residual + relu -> out (stats in sums2[0..1], c = i&63).
// ---------------------------------------------------------------------------
__global__ __launch_bounds__(256) void k_final(
    const float* __restrict__ y, const float* __restrict__ w,
    const float* __restrict__ b, const float* __restrict__ sums2,
    const float* __restrict__ xid, float* __restrict__ out, int total) {
  int i = blockIdx.x * 256 + threadIdx.x;
  if (i >= total) return;
  float cnt  = (float)total;
  float mean = sums2[0] / cnt;
  float var  = sums2[1] / cnt - mean * mean;
  float inv  = 1.0f / (sqrtf(fmaxf(var, 0.0f)) + LN_EPS);
  int c = i & 63;
  float v = (y[i] - mean) * inv * w[c] + b[c] + xid[i];
  out[i] = (v > 0.0f) ? v : 0.0f;
}

// ---------------------------------------------------------------------------
// Host launcher.
// ---------------------------------------------------------------------------
extern "C" void kernel_launch(void* const* d_in, const int* in_sizes, int n_in,
                              void* d_out, int out_size, void* d_ws, size_t ws_size,
                              hipStream_t stream) {
  const float*     x     = (const float*)d_in[0];
  const long long* ei    = (const long long*)d_in[1];
  const float*     eattr = (const float*)d_in[2];
  const float*     W_l   = (const float*)d_in[3];
  const float*     b_l   = (const float*)d_in[4];
  const float*     W_r   = (const float*)d_in[5];
  const float*     b_r   = (const float*)d_in[6];
  const float*     W_e   = (const float*)d_in[7];
  const float*     att   = (const float*)d_in[8];
  const float*     bias  = (const float*)d_in[9];
  const float*     ln1w  = (const float*)d_in[10];
  const float*     ln1b  = (const float*)d_in[11];
  const float*     Wlin  = (const float*)d_in[12];
  const float*     blin  = (const float*)d_in[13];
  const float*     ln2w  = (const float*)d_in[14];
  const float*     ln2b  = (const float*)d_in[15];

  const int N    = in_sizes[0] / 64;            // 50000
  const int E    = in_sizes[1] / 2;             // 400000
  const int Etot = E + N;                       // self loops appended
  const int Npad = ((N + 127) / 128) * 128;     // GEMM row padding

  // Workspace carve-out (256B aligned).
  char*  ws  = (char*)d_ws;
  size_t off = 0;
  auto carve = [&](size_t bytes) -> char* {
    char* p = ws + off;
    off = (off + bytes + 255) & ~(size_t)255;
    return p;
  };
  float*     xl     = (float*)carve((size_t)Npad * 128 * 4);
  float*     xr     = (float*)carve((size_t)Npad * 128 * 4);
  float*     gat    = (float*)carve((size_t)N * 128 * 4);
  float*     hbuf   = (float*)carve((size_t)Npad * 128 * 4);
  float*     ybuf   = (float*)carve((size_t)Npad * 64 * 4);
  float*     logits = (float*)carve((size_t)Etot * 2 * 4);
  unsigned*  nmax   = (unsigned*)carve((size_t)N * 2 * 4);
  float*     denom  = (float*)carve((size_t)N * 2 * 4);
  float*     sums   = (float*)carve(4 * 4);           // [s1, ss1, s2, ss2]
  _Float16*  Wlp    = (_Float16*)carve(8192 * 2);
  _Float16*  Wrp    = (_Float16*)carve(8192 * 2);
  _Float16*  Wlinp  = (_Float16*)carve(8192 * 2);
  (void)ws_size; (void)n_in; (void)out_size;

  const int packBlocks = (8192 + 255) / 256;
  k_pack_b<<<packBlocks, 256, 0, stream>>>(W_l, Wlp, 64, 128);
  k_pack_b<<<packBlocks, 256, 0, stream>>>(W_r, Wrp, 64, 128);
  k_pack_b<<<packBlocks, 256, 0, stream>>>(Wlin, Wlinp, 128, 64);

  k_init<<<(N * 128 + 255) / 256, 256, 0, stream>>>(gat, denom, nmax, sums, N);

  k_gemm_xlxr<<<Npad / 128, 256, 0, stream>>>(x, Wlp, Wrp, b_l, b_r, xl, xr, N);

  const int egrid = (Etot * 2 + 255) / 256;
  k_logits <<<egrid, 256, 0, stream>>>(xl, xr, ei, eattr, W_e, att, logits, nmax, E, N);
  k_expsum <<<egrid, 256, 0, stream>>>(ei, logits, nmax, denom, E, N);
  k_scatter<<<egrid, 256, 0, stream>>>(ei, logits, denom, xl, gat, E, N);

  k_biasstats<<<512, 256, 0, stream>>>(gat, bias, N * 128, sums);
  k_ln1relu<<<(Npad * 128 + 255) / 256, 256, 0, stream>>>(gat, ln1w, ln1b, sums, hbuf,
                                                          N * 128, Npad * 128);

  k_gemm_lin<<<Npad / 128, 256, 0, stream>>>(hbuf, Wlinp, blin, ybuf, Npad);

  k_stats<<<512, 256, 0, stream>>>(ybuf, N * 64, sums + 2);
  k_final<<<(N * 64 + 255) / 256, 256, 0, stream>>>(ybuf, ln2w, ln2b, sums + 2, x,
                                                    (float*)d_out, N * 64);
}
